// MultiHeadAttention_70746701300419
// MI455X (gfx1250) — matrix operand order repaired and checked by reading, not compiled, not run on hardware
//
#include <hip/hip_runtime.h>
#include <hip/hip_bf16.h>

// ---------------------------------------------------------------------------
// MI455X / gfx1250 multi-head causal attention, bf16 WMMA (16x16x32), wave32.
// B=4, N=2048, D=1024, H=16, HD=64.
// ---------------------------------------------------------------------------

typedef __attribute__((ext_vector_type(16))) __bf16 v16bf;
typedef __attribute__((ext_vector_type(8)))  float  v8f;

static constexpr int Bsz = 4, Nseq = 2048, Dm = 1024, Hh = 16, HDd = 64;

__device__ __forceinline__ v8f vzero8() {
    v8f z;
#pragma unroll
    for (int i = 0; i < 8; ++i) z[i] = 0.f;
    return z;
}

union FragU {
    unsigned int u[8];
    uint4        q[2];
    v16bf        v;
};

// A fragment, 16(M) x 32(K) bf16, row-major source with leading dim `ld`.
// Contiguous k order (same as B): lanes 0-15 -> M=lane, K = 0..15; lanes 16-31 -> K = 16..31.
__device__ __forceinline__ v16bf load_a16x32(const __bf16* base, int ld, int m0, int k0, int lane) {
    const int row = m0 + (lane & 15);
    const int kh  = k0 + (lane & 16);              // +16 halves for upper half-wave
    const __bf16* p = base + (size_t)row * ld + kh;
    FragU f;
    f.q[0] = *(const uint4*)(p);                   // K = kh .. kh+7
    f.q[1] = *(const uint4*)(p + 8);               // K = kh+8 .. kh+15
    return f.v;
}

// B fragment, 32(K) x 16(N) bf16 from K-contiguous ("column-major" / pre-transposed)
// storage Bt[n][k] with leading dim `ld`.
// Contiguous k order (same as A): lanes 0-15 -> N=lane, K=0..15; lanes 16-31 -> K=16..31.
__device__ __forceinline__ v16bf load_b32x16(const __bf16* base, int ld, int n0, int k0, int lane) {
    const int row = n0 + (lane & 15);
    const int kh  = k0 + (lane & 16);              // +16 halves for upper half-wave
    const __bf16* p = base + (size_t)row * ld + kh;
    FragU f;
    f.q[0] = *(const uint4*)(p);                   // K = kh .. kh+7
    f.q[1] = *(const uint4*)(p + 8);               // K = kh+8 .. kh+15
    return f.v;
}

__device__ __forceinline__ v8f wmma_bf16(v16bf a, v16bf b, v8f c) {
    return __builtin_amdgcn_wmma_f32_16x16x32_bf16(false, a, false, b, (short)0, c, false, false);
}

// ---------------------------------------------------------------------------
// vectorized f32 -> bf16 (8 elems/thread: 2x b128 in, 1x b128 out)
// ---------------------------------------------------------------------------
__global__ __launch_bounds__(256) void cvt_f32_to_bf16_v8(const float* __restrict__ in,
                                                          __bf16* __restrict__ out, int n8) {
    int i = blockIdx.x * 256 + threadIdx.x;
    if (i >= n8) return;
    const float4 a = *(const float4*)(in + (size_t)i * 8);
    const float4 b = *(const float4*)(in + (size_t)i * 8 + 4);
    union { __bf16 h[8]; uint4 u; } pk;
    pk.h[0] = (__bf16)a.x; pk.h[1] = (__bf16)a.y; pk.h[2] = (__bf16)a.z; pk.h[3] = (__bf16)a.w;
    pk.h[4] = (__bf16)b.x; pk.h[5] = (__bf16)b.y; pk.h[6] = (__bf16)b.z; pk.h[7] = (__bf16)b.w;
    *(uint4*)(out + (size_t)i * 8) = pk.u;
}

// W[k][n] (K x N, f32) -> Wt[n][k] (N x K, bf16), LDS-tiled 32x32 for coalescing
__global__ __launch_bounds__(256) void transpose_cvt_bf16(const float* __restrict__ W,
                                                          __bf16* __restrict__ Wt, int K, int N) {
    __shared__ float tile[32][33];
    const int tx = threadIdx.x;        // 0..31
    const int ty = threadIdx.y;        // 0..7
    const int k0 = blockIdx.x * 32;    // source row block
    const int n0 = blockIdx.y * 32;    // source col block
#pragma unroll
    for (int i = 0; i < 4; ++i) {
        const int r = ty + i * 8;
        tile[r][tx] = W[(size_t)(k0 + r) * N + n0 + tx];
    }
    __syncthreads();
#pragma unroll
    for (int i = 0; i < 4; ++i) {
        const int r = ty + i * 8;
        Wt[(size_t)(n0 + r) * K + k0 + tx] = (__bf16)tile[tx][r];
    }
}

// ---------------------------------------------------------------------------
// bf16 WMMA GEMM, 32x32 tile per wave (4 accumulators, 2 A + 2 B frags per
// k-step -> 4 WMMAs): out = (A @ Bt^T + bias[n]) * scale
// MODE 0: bf16 row-major store    (Q, K projections)
// MODE 1: bf16 transposed-per-head store -> Vt[b][h][d][n]   (V projection)
// MODE 2: f32 row-major store     (output projection)
// ---------------------------------------------------------------------------
template <int MODE>
__global__ __launch_bounds__(256) void gemm_bf16_wmma(const __bf16* __restrict__ A,
                                                      const __bf16* __restrict__ Bt,
                                                      const float* __restrict__ bias,
                                                      void* __restrict__ out,
                                                      int M, int N, int K, float scale) {
    const int lane  = threadIdx.x & 31;
    const int wave  = blockIdx.x * (blockDim.x >> 5) + (threadIdx.x >> 5);
    const int ntile = N >> 5;
    const int tm = wave / ntile;
    const int tn = wave - tm * ntile;
    if (tm >= (M >> 5)) return;
    const int m0 = tm << 5, n0 = tn << 5;

    v8f acc[4] = {vzero8(), vzero8(), vzero8(), vzero8()};
#pragma unroll 2
    for (int k0 = 0; k0 < K; k0 += 32) {
        v16bf a0 = load_a16x32(A, K, m0, k0, lane);
        v16bf a1 = load_a16x32(A, K, m0 + 16, k0, lane);
        v16bf b0 = load_b32x16(Bt, K, n0, k0, lane);
        v16bf b1 = load_b32x16(Bt, K, n0 + 16, k0, lane);
        acc[0] = wmma_bf16(a0, b0, acc[0]);
        acc[1] = wmma_bf16(a0, b1, acc[1]);
        acc[2] = wmma_bf16(a1, b0, acc[2]);
        acc[3] = wmma_bf16(a1, b1, acc[3]);
    }

#pragma unroll
    for (int mi = 0; mi < 2; ++mi) {
#pragma unroll
        for (int ni = 0; ni < 2; ++ni) {
            const v8f& a = acc[mi * 2 + ni];
            const int sm0   = m0 + 16 * mi;
            const int col   = n0 + 16 * ni + (lane & 15);
            const int rbase = sm0 + ((lane & 16) >> 1);
            const float bv  = bias[col];
            if (MODE == 0) {
                __bf16* O = (__bf16*)out;
#pragma unroll
                for (int r = 0; r < 8; ++r)
                    O[(size_t)(rbase + r) * N + col] = (__bf16)((a[r] + bv) * scale);
            } else if (MODE == 1) {
                // Vt[b][h][d][q] : lane's 8 rows are consecutive q -> one b128 store
                __bf16* O   = (__bf16*)out;
                const int bb = sm0 >> 11;                       // row / Nseq
                const int qi = (sm0 & (Nseq - 1)) + ((lane & 16) >> 1);
                const int hh = col >> 6, dd = col & 63;
                union { __bf16 h[8]; uint4 u; } pk;
#pragma unroll
                for (int r = 0; r < 8; ++r) pk.h[r] = (__bf16)((a[r] + bv) * scale);
                *(uint4*)(O + ((((size_t)bb * Hh + hh) * HDd + dd) * Nseq + qi)) = pk.u;
            } else {
                float* O = (float*)out;
#pragma unroll
                for (int r = 0; r < 8; ++r)
                    O[(size_t)(rbase + r) * N + col] = (a[r] + bv) * scale;
            }
        }
    }
}

// ---------------------------------------------------------------------------
// LDS-free causal flash attention (transposed formulation), one wave per
// (b, h, 16-query tile).  S^T = K*Q^T ; O^T = V^T * P^T.
// Q is pre-scaled by 1/sqrt(HD). Output written bf16 at [b][q][h][d] (= [B*N, D]).
// ---------------------------------------------------------------------------
__global__ __launch_bounds__(256) void flash_attn_wmma(const __bf16* __restrict__ Q,
                                                       const __bf16* __restrict__ Kmat,
                                                       const __bf16* __restrict__ Vt,
                                                       __bf16* __restrict__ Oout) {
    const int lane   = threadIdx.x & 31;
    const int wave   = blockIdx.x * (blockDim.x >> 5) + (threadIdx.x >> 5);
    const int qtiles = Nseq >> 4;
    const int b   = wave / (Hh * qtiles);
    if (b >= Bsz) return;
    const int rem = wave - b * (Hh * qtiles);
    const int h   = rem / qtiles;
    const int qt  = rem - h * qtiles;
    const int q0  = qt << 4;

    const __bf16* Qb = Q    + (size_t)b * Nseq * Dm + h * HDd;   // row stride Dm
    const __bf16* Kb = Kmat + (size_t)b * Nseq * Dm + h * HDd;   // row stride Dm
    const __bf16* Vb = Vt   + ((size_t)b * Hh + h) * HDd * Nseq; // row stride Nseq

    // Q^T as B operand (d x 16 queries), loaded once: d 0..31 and 32..63
    const v16bf qf0 = load_b32x16(Qb, Dm, q0, 0, lane);
    const v16bf qf1 = load_b32x16(Qb, Dm, q0, 32, lane);

    float m_i = -1e30f, l_i = 0.f;
    v8f o[4] = {vzero8(), vzero8(), vzero8(), vzero8()};

    const int qcol  = q0 + (lane & 15);       // this lane's query column
    const int kbase = (lane & 16) >> 1;       // key-row offset of this half-wave
    const int qmax  = q0 + 15;

    for (int key0 = 0; key0 <= qmax; key0 += 32) {
        if (key0 + 32 <= qmax) {              // prefetch next K tile (global_prefetch_b8)
            __builtin_prefetch(Kb + (size_t)(key0 + 32 + (lane & 15)) * Dm, 0, 0);
            __builtin_prefetch(Kb + (size_t)(key0 + 48 + (lane & 15)) * Dm, 0, 0);
        }
        // --- S^T tiles: keys [key0..+15] and [key0+16..+31] x 16 queries ---
        v8f s0 = vzero8(), s1 = vzero8();
        {
            v16bf ka = load_a16x32(Kb, Dm, key0, 0, lane);
            v16bf kc = load_a16x32(Kb, Dm, key0, 32, lane);
            s0 = wmma_bf16(ka, qf0, s0);
            s0 = wmma_bf16(kc, qf1, s0);
        }
        {
            v16bf ka = load_a16x32(Kb, Dm, key0 + 16, 0, lane);
            v16bf kc = load_a16x32(Kb, Dm, key0 + 16, 32, lane);
            s1 = wmma_bf16(ka, qf0, s1);
            s1 = wmma_bf16(kc, qf1, s1);
        }
        // --- causal mask + per-query running max ---
        float mx = -1e30f;
#pragma unroll
        for (int r = 0; r < 8; ++r) {
            const int kg = key0 + kbase + r;
            float v0 = (kg > qcol) ? -1e30f : s0[r];
            float v1 = (kg + 16 > qcol) ? -1e30f : s1[r];
            s0[r] = v0; s1[r] = v1;
            mx = fmaxf(mx, fmaxf(v0, v1));
        }
        mx = fmaxf(mx, __shfl_xor(mx, 16, 32));       // combine key halves
        const float mnew  = fmaxf(m_i, mx);
        const float alpha = __expf(m_i - mnew);

        // --- P^T = exp(S^T - m), packed bf16 pairs per VGPR ---
        float psum = 0.f;
        unsigned int p0[4], p1[4];
#pragma unroll
        for (int i = 0; i < 4; ++i) {
            float a0 = __expf(s0[2 * i]     - mnew);
            float a1 = __expf(s0[2 * i + 1] - mnew);
            float b0 = __expf(s1[2 * i]     - mnew);
            float b1 = __expf(s1[2 * i + 1] - mnew);
            psum += a0 + a1 + b0 + b1;
            union { __bf16 h[2]; unsigned int u; } c0, c1;
            c0.h[0] = (__bf16)a0; c0.h[1] = (__bf16)a1; p0[i] = c0.u;
            c1.h[0] = (__bf16)b0; c1.h[1] = (__bf16)b1; p1[i] = c1.u;
        }
        psum += __shfl_xor(psum, 16, 32);
        l_i = l_i * alpha + psum;
        m_i = mnew;

        // --- build P^T B-fragment (32 keys x 16 queries): swap key halves ---
        FragU pb;
        const bool lo = (lane & 16) == 0;
#pragma unroll
        for (int i = 0; i < 4; ++i) {
            unsigned int x0 = __shfl_xor(p0[i], 16, 32);
            unsigned int x1 = __shfl_xor(p1[i], 16, 32);
            pb.u[i]     = lo ? p0[i] : x1;   // K = 0..15   (lanes<16) / 16..31 (lanes>=16)
            pb.u[4 + i] = lo ? x0    : p1[i];
        }

        // --- O^T += V^T * P^T, 4 d-chunks of 16 ---
#pragma unroll
        for (int c = 0; c < 4; ++c) {
            v16bf va = load_a16x32(Vb, Nseq, c * 16, key0, lane);
            v8f cc;
#pragma unroll
            for (int r = 0; r < 8; ++r) cc[r] = o[c][r] * alpha;
            o[c] = wmma_bf16(va, pb.v, cc);
        }
    }

    // --- normalize + store: lane's 8 rows are 8 consecutive d columns -> b128 ---
    const float inv = 1.f / l_i;
    const size_t obase = ((size_t)b * Nseq + q0 + (lane & 15)) * Dm + h * HDd + ((lane & 16) >> 1);
#pragma unroll
    for (int c = 0; c < 4; ++c) {
        union { __bf16 h[8]; uint4 u; } pk;
#pragma unroll
        for (int r = 0; r < 8; ++r) pk.h[r] = (__bf16)(o[c][r] * inv);
        *(uint4*)(Oout + obase + c * 16) = pk.u;
    }
}

// ---------------------------------------------------------------------------
extern "C" void kernel_launch(void* const* d_in, const int* in_sizes, int n_in,
                              void* d_out, int out_size, void* d_ws, size_t ws_size,
                              hipStream_t stream) {
    (void)in_sizes; (void)n_in; (void)out_size; (void)ws_size;
    const float* x  = (const float*)d_in[0];
    const float* Wq = (const float*)d_in[1];
    const float* bq = (const float*)d_in[2];
    const float* Wk = (const float*)d_in[3];
    const float* bk = (const float*)d_in[4];
    const float* Wv = (const float*)d_in[5];
    const float* bv = (const float*)d_in[6];
    const float* Wo = (const float*)d_in[7];
    const float* bo = (const float*)d_in[8];

    const int M = Bsz * Nseq;            // 8192
    const int D = Dm;                    // 1024

    char*  ws  = (char*)d_ws;
    size_t off = 0;
    auto take = [&](size_t bytes) -> void* {
        void* p = ws + off;
        off += (bytes + 255) & ~(size_t)255;
        return p;
    };
    __bf16* xb   = (__bf16*)take((size_t)M * D * 2);
    __bf16* wqt  = (__bf16*)take((size_t)D * D * 2);
    __bf16* wkt  = (__bf16*)take((size_t)D * D * 2);
    __bf16* wvt  = (__bf16*)take((size_t)D * D * 2);
    __bf16* wot  = (__bf16*)take((size_t)D * D * 2);
    __bf16* qb   = (__bf16*)take((size_t)M * D * 2);
    __bf16* kb   = (__bf16*)take((size_t)M * D * 2);
    __bf16* vt   = (__bf16*)take((size_t)M * D * 2);   // [B,H,HD,N]
    __bf16* attn = (__bf16*)take((size_t)M * D * 2);   // [B,N,H,HD] = [M,D]

    // --- convert inputs to bf16 ---
    cvt_f32_to_bf16_v8<<<(M * D / 8 + 255) / 256, 256, 0, stream>>>(x, xb, M * D / 8);
    {
        dim3 tgrid(D / 32, D / 32), tblk(32, 8);
        transpose_cvt_bf16<<<tgrid, tblk, 0, stream>>>(Wq, wqt, D, D);
        transpose_cvt_bf16<<<tgrid, tblk, 0, stream>>>(Wk, wkt, D, D);
        transpose_cvt_bf16<<<tgrid, tblk, 0, stream>>>(Wv, wvt, D, D);
        transpose_cvt_bf16<<<tgrid, tblk, 0, stream>>>(Wo, wot, D, D);
    }

    // --- projections: (M/32)*(D/32) 32x32 tiles, 8 waves/block ---
    const int tiles  = (M >> 5) * (D >> 5);   // 8192
    const int blocks = tiles / 8;             // 1024
    gemm_bf16_wmma<0><<<blocks, 256, 0, stream>>>(xb, wqt, bq, qb, M, D, D, 0.125f); // q/sqrt(64)
    gemm_bf16_wmma<0><<<blocks, 256, 0, stream>>>(xb, wkt, bk, kb, M, D, D, 1.0f);
    gemm_bf16_wmma<1><<<blocks, 256, 0, stream>>>(xb, wvt, bv, vt, M, D, D, 1.0f);

    // --- causal flash attention: B*H*(N/16) = 8192 waves ---
    flash_attn_wmma<<<(Bsz * Hh * (Nseq >> 4)) / 8, 256, 0, stream>>>(qb, kb, vt, attn);

    // --- output projection, f32 result ---
    gemm_bf16_wmma<2><<<blocks, 256, 0, stream>>>(attn, wot, bo, (float*)d_out, M, D, D, 1.0f);
}
